// Net_62801011802909
// MI455X (gfx1250) — compile-verified, matrix operand order checked by reading, not run on hardware
//
#include <hip/hip_runtime.h>
#include <hip/hip_bf16.h>
#include <stdint.h>

// ---------------- types ----------------
typedef __attribute__((ext_vector_type(16))) __bf16 bf16x16;
typedef __attribute__((ext_vector_type(8)))  float  f32x8;

#define Bq    1024
#define TIN   256
#define TOUT  64
#define INDIM 33
#define EMB   32
#define HID   256
#define G3    (3 * HID)   // 768

__device__ __forceinline__ f32x8 wmma_bf16(bf16x16 a, bf16x16 b, f32x8 c) {
  return __builtin_amdgcn_wmma_f32_16x16x32_bf16(false, a, false, b, (short)0, c, false, false);
}

// Opaque integer: breaks loop-invariance of derived addresses (prevents LICM
// hoist+spill of per-step weight fragments) WITHOUT obscuring the pointer's
// provenance, so InferAddressSpaces still emits global_load_b128 (not flat).
__device__ __forceinline__ int opaque_i(int v) {
  asm volatile("" : "+v"(v));
  return v;
}

__device__ __forceinline__ float sigf(float x) {
  return 1.0f / (1.0f + __expf(-x));
}
__device__ __forceinline__ float tanhfast(float x) {
  float e = __expf(-2.0f * x);
  return (1.0f - e) / (1.0f + e);
}

// A fragment: 16 rows x 32 k, bf16, row-major with leading dim ld (elements).
__device__ __forceinline__ bf16x16 load_frag_A(const __bf16* base, int ld) {
  int lane = threadIdx.x & 31;
  int row  = lane & 15;
  int koff = (lane >> 4) * 16;
  return *(const bf16x16*)(base + row * ld + koff);
}

// B fragment for D = A * W^T : B[k,n] = W[n,k]; W row-major [N_total, ldk] bf16.
__device__ __forceinline__ bf16x16 load_frag_B(const __bf16* W, int ldk, int n0, int k0) {
  int lane = threadIdx.x & 31;
  int col  = lane & 15;
  int koff = (lane >> 4) * 16;
  return *(const bf16x16*)(W + (size_t)(n0 + col) * ldk + k0 + koff);
}

// ---------------- f32 -> bf16 conversion ----------------
__global__ void cvt_bf16_kernel(const float* __restrict__ src, __bf16* __restrict__ dst, int n) {
  int i = blockIdx.x * blockDim.x + threadIdx.x;
  int stride = gridDim.x * blockDim.x;
  for (; i < n; i += stride) dst[i] = (__bf16)src[i];
}

// ---------------- embedder MLP: 33 -> 32 -> 32 -> 32, out bf16 ----------------
__global__ void embed_kernel(const float* __restrict__ in_seq,
                             const float* __restrict__ ew1, const float* __restrict__ eb1,
                             const float* __restrict__ ew2, const float* __restrict__ eb2,
                             const float* __restrict__ ew3, const float* __restrict__ eb3,
                             __bf16* __restrict__ emb) {
  __shared__ float w1s[32 * 33];
  __shared__ float w2s[32 * 32];
  __shared__ float w3s[32 * 32];
  __shared__ float b1s[32], b2s[32], b3s[32];
  __shared__ float xin[8][33];
  __shared__ float h1[8][32];
  __shared__ float h2[8][32];

  int tid = threadIdx.x;
  for (int i = tid; i < 32 * 33; i += 256) w1s[i] = ew1[i];
  for (int i = tid; i < 32 * 32; i += 256) w2s[i] = ew2[i];
  for (int i = tid; i < 32 * 32; i += 256) w3s[i] = ew3[i];
  if (tid < 32) { b1s[tid] = eb1[tid]; b2s[tid] = eb2[tid]; b3s[tid] = eb3[tid]; }

  int sub  = tid >> 5;
  int lane = tid & 31;
  size_t row = (size_t)blockIdx.x * 8 + sub;

  for (int k = lane; k < INDIM; k += 32) xin[sub][k] = in_seq[row * INDIM + k];
  __syncthreads();

  float a = b1s[lane];
  #pragma unroll
  for (int k = 0; k < INDIM; ++k) a += w1s[lane * INDIM + k] * xin[sub][k];
  h1[sub][lane] = fmaxf(a, 0.0f);
  __syncthreads();

  float b = b2s[lane];
  #pragma unroll
  for (int k = 0; k < 32; ++k) b += w2s[lane * 32 + k] * h1[sub][k];
  h2[sub][lane] = fmaxf(b, 0.0f);
  __syncthreads();

  float c = b3s[lane];
  #pragma unroll
  for (int k = 0; k < 32; ++k) c += w3s[lane * 32 + k] * h2[sub][k];
  emb[row * EMB + lane] = (__bf16)c;
}

// ---------------- warm GRU: persistent per 32-row batch tile ----------------
// grid = 32 blocks x 512 threads (16 waves). Wave w owns gate-column tile w.
__global__ void __launch_bounds__(512)
warm_gru_kernel(const __bf16* __restrict__ emb,
                const int*    __restrict__ lengths_in,
                const float*  __restrict__ g_bi,
                const float*  __restrict__ g_bh,
                const __bf16* __restrict__ wi_bf,       // [768, 32]
                const __bf16* __restrict__ wh_bf,       // [768, 256]
                float* __restrict__ hidden) {           // [B, HID]
  __shared__ __align__(32) float  h_f32[32][HID];
  __shared__ __align__(32) __bf16 h_bf [32][HID];
  __shared__ __align__(32) __bf16 x_bf [32][EMB];
  __shared__ float bi_s[G3], bh_s[G3];
  __shared__ int   len_s[32];

  int tid = threadIdx.x;
  int b0  = blockIdx.x * 32;

  for (int i = tid; i < 32 * HID; i += 512) {
    ((float*)h_f32)[i]  = 0.0f;
    ((__bf16*)h_bf)[i]  = (__bf16)0.0f;
  }
  for (int i = tid; i < G3; i += 512) { bi_s[i] = g_bi[i]; bh_s[i] = g_bh[i]; }
  if (tid < 32) len_s[tid] = lengths_in[b0 + tid];
  __syncthreads();

  int wv    = tid >> 5;
  int lane  = tid & 31;
  int colL  = lane & 15;
  int rhalf = (lane >> 4) * 8;
  int cg    = wv * 16;
  int col   = cg + colL;

  // xi B-fragments are small (24 VGPRs): keep register-resident across time.
  bf16x16 bwr = load_frag_B(wi_bf, EMB, 0 * HID + cg, 0);
  bf16x16 bwz = load_frag_B(wi_bf, EMB, 1 * HID + cg, 0);
  bf16x16 bwn = load_frag_B(wi_bf, EMB, 2 * HID + cg, 0);

  for (int t = 0; t < TIN; ++t) {
    // opaque zero offset: keeps Wh fragment loads inside the loop (no
    // hoist+spill) while preserving global address-space inference
    const __bf16* whp = wh_bf + opaque_i(0);

    {
      // stage x_t tile [32 x 32] bf16 as packed 32-bit copies
      int m  = tid >> 4;           // 512 threads -> 32 rows x 16 dwords
      int kw = tid & 15;
      const uint32_t* src32 =
          (const uint32_t*)(emb + ((size_t)(b0 + m) * TIN + t) * EMB);
      ((uint32_t*)&x_bf[m][0])[kw] = src32[kw];
    }
    __syncthreads();

    f32x8 accR[2], accZ[2], accNX[2], accNH[2];
    #pragma unroll
    for (int mt = 0; mt < 2; ++mt) {
      bf16x16 ax = load_frag_A(&x_bf[mt * 16][0], EMB);
      f32x8 z = {};
      accR[mt]  = wmma_bf16(ax, bwr, z);
      accZ[mt]  = wmma_bf16(ax, bwz, z);
      accNX[mt] = wmma_bf16(ax, bwn, z);
      accNH[mt] = z;
    }
    #pragma unroll
    for (int k = 0; k < HID; k += 32) {
      bf16x16 a0 = load_frag_A(&h_bf[0][k],  HID);
      bf16x16 a1 = load_frag_A(&h_bf[16][k], HID);
      bf16x16 br = load_frag_B(whp, HID, 0 * HID + cg, k);
      accR[0] = wmma_bf16(a0, br, accR[0]);
      accR[1] = wmma_bf16(a1, br, accR[1]);
      bf16x16 bz = load_frag_B(whp, HID, 1 * HID + cg, k);
      accZ[0] = wmma_bf16(a0, bz, accZ[0]);
      accZ[1] = wmma_bf16(a1, bz, accZ[1]);
      bf16x16 bn = load_frag_B(whp, HID, 2 * HID + cg, k);
      accNH[0] = wmma_bf16(a0, bn, accNH[0]);
      accNH[1] = wmma_bf16(a1, bn, accNH[1]);
    }
    __syncthreads();   // all waves done reading h_bf before rewrite

    float biR  = bi_s[col]       + bh_s[col];
    float biZ  = bi_s[HID + col] + bh_s[HID + col];
    float biNX = bi_s[2 * HID + col];
    float bhN  = bh_s[2 * HID + col];
    #pragma unroll
    for (int mt = 0; mt < 2; ++mt) {
      #pragma unroll
      for (int i = 0; i < 8; ++i) {
        int m = mt * 16 + rhalf + i;
        float r  = sigf(accR[mt][i] + biR);
        float zg = sigf(accZ[mt][i] + biZ);
        float nv = tanhfast(accNX[mt][i] + biNX + r * (accNH[mt][i] + bhN));
        float hold = h_f32[m][col];
        float hnew = (1.0f - zg) * nv + zg * hold;
        hnew = (t < len_s[m]) ? hnew : hold;
        h_f32[m][col] = hnew;
        h_bf[m][col]  = (__bf16)hnew;
      }
    }
    __syncthreads();
  }

  for (int i = tid; i < 32 * HID; i += 512) {
    int m = i >> 8, k = i & 255;
    hidden[(size_t)(b0 + m) * HID + k] = h_f32[m][k];
  }
}

// ---------------- decode: GRUCell + MLP, persistent per 32-row tile ----------------
__global__ void __launch_bounds__(512)
decode_kernel(const float* __restrict__ hidden,
              const int*   __restrict__ lengths_out,
              const float* __restrict__ last_cords,
              const float* __restrict__ c_wi,         // [768, 2]
              const float* __restrict__ c_bi,
              const float* __restrict__ c_bh,
              const __bf16* __restrict__ cwh_bf,      // [768, 256]
              const __bf16* __restrict__ pw1_bf,      // [64, 256]
              const float* __restrict__ p_b1,
              const __bf16* __restrict__ pw2_bf,      // [64, 64]
              const float* __restrict__ p_b2,
              const float* __restrict__ p_w3,         // [2, 64]
              const float* __restrict__ p_b3,
              float* __restrict__ preds) {            // [B, TOUT, 2]
  __shared__ __align__(32) float  h_f32[32][HID];
  __shared__ __align__(32) __bf16 h_bf [32][HID];
  __shared__ __align__(32) __bf16 y1_bf[32][64];
  __shared__ __align__(32) float  y2_s [32][64];
  __shared__ float cwi_s[G3 * 2];
  __shared__ float cbi_s[G3], cbh_s[G3];
  __shared__ float lc_s[32][2];
  __shared__ int   len_s[32];

  int tid = threadIdx.x;
  int b0  = blockIdx.x * 32;

  for (int i = tid; i < 32 * HID; i += 512) {
    int m = i >> 8, k = i & 255;
    float v = hidden[(size_t)(b0 + m) * HID + k];
    h_f32[m][k] = v;
    h_bf[m][k]  = (__bf16)v;
  }
  for (int i = tid; i < G3; i += 512) {
    cbi_s[i] = c_bi[i]; cbh_s[i] = c_bh[i];
    cwi_s[2 * i] = c_wi[2 * i]; cwi_s[2 * i + 1] = c_wi[2 * i + 1];
  }
  if (tid < 64) { int m = tid >> 1, c = tid & 1; lc_s[m][c] = last_cords[(size_t)(b0 + m) * 2 + c]; }
  if (tid < 32) len_s[tid] = lengths_out[b0 + tid];
  __syncthreads();

  int wv    = tid >> 5;
  int lane  = tid & 31;
  int colL  = lane & 15;
  int rhalf = (lane >> 4) * 8;
  int cg    = wv * 16;
  int col   = cg + colL;

  for (int t = 0; t < TOUT; ++t) {
    const __bf16* cwhp = cwh_bf + opaque_i(0);

    // ---- GRU cell: hh = h @ c_wh^T via WMMA ----
    f32x8 accR[2] = {}, accZ[2] = {}, accNH[2] = {};
    #pragma unroll
    for (int k = 0; k < HID; k += 32) {
      bf16x16 a0 = load_frag_A(&h_bf[0][k],  HID);
      bf16x16 a1 = load_frag_A(&h_bf[16][k], HID);
      bf16x16 br = load_frag_B(cwhp, HID, 0 * HID + cg, k);
      accR[0] = wmma_bf16(a0, br, accR[0]);
      accR[1] = wmma_bf16(a1, br, accR[1]);
      bf16x16 bz = load_frag_B(cwhp, HID, 1 * HID + cg, k);
      accZ[0] = wmma_bf16(a0, bz, accZ[0]);
      accZ[1] = wmma_bf16(a1, bz, accZ[1]);
      bf16x16 bn = load_frag_B(cwhp, HID, 2 * HID + cg, k);
      accNH[0] = wmma_bf16(a0, bn, accNH[0]);
      accNH[1] = wmma_bf16(a1, bn, accNH[1]);
    }
    __syncthreads();

    float wiR0 = cwi_s[2 * col],             wiR1 = cwi_s[2 * col + 1];
    float wiZ0 = cwi_s[2 * (HID + col)],     wiZ1 = cwi_s[2 * (HID + col) + 1];
    float wiN0 = cwi_s[2 * (2 * HID + col)], wiN1 = cwi_s[2 * (2 * HID + col) + 1];
    float biR = cbi_s[col],           bhR = cbh_s[col];
    float biZ = cbi_s[HID + col],     bhZ = cbh_s[HID + col];
    float biN = cbi_s[2 * HID + col], bhN = cbh_s[2 * HID + col];

    #pragma unroll
    for (int mt = 0; mt < 2; ++mt) {
      #pragma unroll
      for (int i = 0; i < 8; ++i) {
        int m = mt * 16 + rhalf + i;
        float l0 = lc_s[m][0], l1 = lc_s[m][1];
        float r  = sigf(l0 * wiR0 + l1 * wiR1 + biR + accR[mt][i] + bhR);
        float zg = sigf(l0 * wiZ0 + l1 * wiZ1 + biZ + accZ[mt][i] + bhZ);
        float nv = tanhfast(l0 * wiN0 + l1 * wiN1 + biN + r * (accNH[mt][i] + bhN));
        float hnew = (1.0f - zg) * nv + zg * h_f32[m][col];
        h_f32[m][col] = hnew;             // forward value unaffected by detach blend
        h_bf[m][col]  = (__bf16)hnew;
      }
    }
    __syncthreads();

    // ---- MLP layer 1: y1 = relu(h @ p_w1^T + b1), [32,256] x [256,64] ----
    if (wv < 8) {
      const __bf16* p1 = pw1_bf + opaque_i(0);
      int mt = wv & 1, nt = wv >> 1;
      f32x8 acc = {};
      #pragma unroll
      for (int k = 0; k < HID; k += 32) {
        bf16x16 a = load_frag_A(&h_bf[mt * 16][k], HID);
        bf16x16 b = load_frag_B(p1, HID, nt * 16, k);
        acc = wmma_bf16(a, b, acc);
      }
      int oc = nt * 16 + colL;
      float bb = p_b1[oc];
      #pragma unroll
      for (int i = 0; i < 8; ++i) {
        int m = mt * 16 + rhalf + i;
        y1_bf[m][oc] = (__bf16)fmaxf(acc[i] + bb, 0.0f);
      }
    }
    __syncthreads();

    // ---- MLP layer 2: y2 = relu(y1 @ p_w2^T + b2), [32,64] x [64,64] ----
    if (wv < 8) {
      const __bf16* p2 = pw2_bf + opaque_i(0);
      int mt = wv & 1, nt = wv >> 1;
      f32x8 acc = {};
      #pragma unroll
      for (int k = 0; k < 64; k += 32) {
        bf16x16 a = load_frag_A(&y1_bf[mt * 16][k], 64);
        bf16x16 b = load_frag_B(p2, 64, nt * 16, k);
        acc = wmma_bf16(a, b, acc);
      }
      int oc = nt * 16 + colL;
      float bb = p_b2[oc];
      #pragma unroll
      for (int i = 0; i < 8; ++i) {
        int m = mt * 16 + rhalf + i;
        y2_s[m][oc] = fmaxf(acc[i] + bb, 0.0f);
      }
    }
    __syncthreads();

    // ---- output head: lc = y2 @ p_w3^T + b3, masked; write preds ----
    if (tid < 64) {
      int m = tid >> 1, c = tid & 1;
      float s = p_b3[c];
      #pragma unroll
      for (int k = 0; k < 64; ++k) s += y2_s[m][k] * p_w3[c * 64 + k];
      float out = (t < len_s[m]) ? s : 0.0f;
      preds[((size_t)(b0 + m) * TOUT + t) * 2 + c] = out;
      lc_s[m][c] = out;
    }
    __syncthreads();
  }
}

// ---------------- host launcher ----------------
extern "C" void kernel_launch(void* const* d_in, const int* in_sizes, int n_in,
                              void* d_out, int out_size, void* d_ws, size_t ws_size,
                              hipStream_t stream) {
  const float* in_seq      = (const float*)d_in[0];
  const int*   lengths_in  = (const int*)  d_in[1];
  const int*   lengths_out = (const int*)  d_in[2];
  const float* last_cords  = (const float*)d_in[3];
  const float* e_w1 = (const float*)d_in[4];
  const float* e_b1 = (const float*)d_in[5];
  const float* e_w2 = (const float*)d_in[6];
  const float* e_b2 = (const float*)d_in[7];
  const float* e_w3 = (const float*)d_in[8];
  const float* e_b3 = (const float*)d_in[9];
  const float* g_wi = (const float*)d_in[10];
  const float* g_wh = (const float*)d_in[11];
  const float* g_bi = (const float*)d_in[12];
  const float* g_bh = (const float*)d_in[13];
  const float* c_wi = (const float*)d_in[14];
  const float* c_wh = (const float*)d_in[15];
  const float* c_bi = (const float*)d_in[16];
  const float* c_bh = (const float*)d_in[17];
  const float* p_w1 = (const float*)d_in[18];
  const float* p_b1 = (const float*)d_in[19];
  const float* p_w2 = (const float*)d_in[20];
  const float* p_b2 = (const float*)d_in[21];
  const float* p_w3 = (const float*)d_in[22];
  const float* p_b3 = (const float*)d_in[23];

  char* ws = (char*)d_ws;
  size_t off = 0;
  __bf16* emb    = (__bf16*)(ws + off); off += (size_t)Bq * TIN * EMB * 2;
  float*  hidden = (float*) (ws + off); off += (size_t)Bq * HID * 4;
  __bf16* wi_bf  = (__bf16*)(ws + off); off += (size_t)G3 * EMB * 2;
  __bf16* wh_bf  = (__bf16*)(ws + off); off += (size_t)G3 * HID * 2;
  __bf16* cwh_bf = (__bf16*)(ws + off); off += (size_t)G3 * HID * 2;
  __bf16* pw1_bf = (__bf16*)(ws + off); off += (size_t)64 * HID * 2;
  __bf16* pw2_bf = (__bf16*)(ws + off); off += (size_t)64 * 64 * 2;

  cvt_bf16_kernel<<<(G3 * EMB + 255) / 256, 256, 0, stream>>>(g_wi, wi_bf, G3 * EMB);
  cvt_bf16_kernel<<<(G3 * HID + 255) / 256, 256, 0, stream>>>(g_wh, wh_bf, G3 * HID);
  cvt_bf16_kernel<<<(G3 * HID + 255) / 256, 256, 0, stream>>>(c_wh, cwh_bf, G3 * HID);
  cvt_bf16_kernel<<<(64 * HID + 255) / 256, 256, 0, stream>>>(p_w1, pw1_bf, 64 * HID);
  cvt_bf16_kernel<<<(64 * 64 + 255) / 256, 256, 0, stream>>>(p_w2, pw2_bf, 64 * 64);

  embed_kernel<<<(Bq * TIN) / 8, 256, 0, stream>>>(in_seq, e_w1, e_b1, e_w2, e_b2,
                                                   e_w3, e_b3, emb);

  warm_gru_kernel<<<Bq / 32, 512, 0, stream>>>(emb, lengths_in, g_bi, g_bh,
                                               wi_bf, wh_bf, hidden);

  decode_kernel<<<Bq / 32, 512, 0, stream>>>(hidden, lengths_out, last_cords,
                                             c_wi, c_bi, c_bh, cwh_bf,
                                             pw1_bf, p_b1, pw2_bf, p_b2,
                                             p_w3, p_b3, (float*)d_out);
}
